// Attention_81157702025592
// MI455X (gfx1250) — matrix operand order repaired and checked by reading, not compiled, not run on hardware
//
#include <hip/hip_runtime.h>
#include <hip/hip_bf16.h>
#include <stdint.h>

// MI455X / gfx1250, wave32. Matrix math via v_wmma_f32_16x16x32_bf16;
// flash-attention tile staging via global_load_async_to_lds_b128 (ASYNCcnt).
// B=2, L=2048, D=4096, H=32, HKV=8, HD=128.

typedef __bf16 bf16;
typedef __attribute__((ext_vector_type(16))) __bf16 v16bf;
typedef __attribute__((ext_vector_type(8)))  float  v8f;
typedef __attribute__((ext_vector_type(4)))  int    v4i;

#define AS3 __attribute__((address_space(3)))
#define AS1 __attribute__((address_space(1)))

union Frag16 {            // 16 bf16 = 8 VGPRs: one WMMA A or B operand
    v16bf v;
    uint4 q[2];
    uint2 u[4];
};

__device__ __forceinline__ unsigned pack_bf16x2(float a, float b) {
    unsigned short lo = __builtin_bit_cast(unsigned short, (bf16)a);
    unsigned short hi = __builtin_bit_cast(unsigned short, (bf16)b);
    return (unsigned)lo | ((unsigned)hi << 16);
}
__device__ __forceinline__ uint2 cvt4(float4 f) {
    uint2 r;
    r.x = pack_bf16x2(f.x, f.y);
    r.y = pack_bf16x2(f.z, f.w);
    return r;
}
__device__ __forceinline__ v8f wmma_bf16(v16bf a, v16bf b, v8f c) {
    return __builtin_amdgcn_wmma_f32_16x16x32_bf16(
        false, a, false, b, (short)0, c, false, false);
}

// Async 16B global -> LDS copy (per-lane addresses), tracked by ASYNCcnt.
// Builtin signature (from hipcc diagnostic): (global int4*, local int4*, Ii, Ii).
__device__ __forceinline__ void async_b128(const bf16* g, bf16* l) {
#if __has_builtin(__builtin_amdgcn_global_load_async_to_lds_b128)
    __builtin_amdgcn_global_load_async_to_lds_b128(
        (AS1 v4i*)(uintptr_t)g,
        (AS3 v4i*)(uint32_t)(uintptr_t)l,
        0, 0);
#else
    asm volatile("global_load_async_to_lds_b128 %0, %1, off"
                 :: "v"((uint32_t)(uintptr_t)l), "v"(g)
                 : "memory");
#endif
}
__device__ __forceinline__ void wait_async0() {
#if __has_builtin(__builtin_amdgcn_s_wait_asynccnt)
    __builtin_amdgcn_s_wait_asynccnt(0);
#else
    asm volatile("s_wait_asynccnt 0x0" ::: "memory");
#endif
}

// ---------------------------------------------------------------------------
// C[M,N] = A[M,K] * W[N,K]^T, fp32 in, fp32/bf16 out, bf16 WMMA inside.
// 128x128 block tile, 256 threads (8 waves), K-step 32, register-prefetch
// software pipeline. Leading dims are template constants -> immediate-offset
// addressing in the epilogue.
// ---------------------------------------------------------------------------
#define GSTRIDE 40   // 32 + 8 bf16 pad (80B rows, 16B aligned)

template<int LDA, int LDW, int LDC, bool OUTBF16>
__global__ __launch_bounds__(256) void gemm_xwT(
    const float* __restrict__ A, const float* __restrict__ W,
    void* __restrict__ Cout, int K)
{
    __shared__ bf16 As[128 * GSTRIDE];
    __shared__ bf16 Ws[128 * GSTRIDE];

    const int tid  = threadIdx.x;
    const int lane = tid & 31;
    const int w    = tid >> 5;
    const int wm   = w >> 2;      // 0..1 -> 64 rows
    const int wn   = w & 3;       // 0..3 -> 32 cols
    const int bm   = blockIdx.y * 128;
    const int bn   = blockIdx.x * 128;
    const int lr   = lane & 15;
    const int half = lane >> 4;

    const v8f zero8 = {0.f,0.f,0.f,0.f,0.f,0.f,0.f,0.f};
    v8f acc[4][2];
    #pragma unroll
    for (int mi = 0; mi < 4; ++mi)
        #pragma unroll
        for (int ni = 0; ni < 2; ++ni) acc[mi][ni] = zero8;

    // global load mapping: thread -> (row band, 16B column chunk)
    const int grow0 = tid >> 3;       // 0..31, +i*32
    const int gcol  = (tid & 7) * 4;  // 0..28
    const float* ga = A + (size_t)(bm + grow0) * LDA + gcol;
    const float* gw = W + (size_t)(bn + grow0) * LDW + gcol;

    float4 ra[4], rw[4];
    #pragma unroll
    for (int i = 0; i < 4; ++i) {
        ra[i] = *(const float4*)(ga + (size_t)i * 32 * LDA);
        rw[i] = *(const float4*)(gw + (size_t)i * 32 * LDW);
    }

    const int akb = half ? 8 : 0;     // ISA 16-bit A-layout K base
    const int bkb = half ? 8 : 0;     // ISA 16-bit B-layout K base

    for (int k0 = 0; k0 < K; k0 += 32) {
        __syncthreads();
        #pragma unroll
        for (int i = 0; i < 4; ++i) {
            *(uint2*)&As[(grow0 + i * 32) * GSTRIDE + gcol] = cvt4(ra[i]);
            *(uint2*)&Ws[(grow0 + i * 32) * GSTRIDE + gcol] = cvt4(rw[i]);
        }
        __syncthreads();

        if (k0 + 32 < K) {            // prefetch next K-step into registers
            #pragma unroll
            for (int i = 0; i < 4; ++i) {
                ra[i] = *(const float4*)(ga + (size_t)i * 32 * LDA + k0 + 32);
                rw[i] = *(const float4*)(gw + (size_t)i * 32 * LDW + k0 + 32);
            }
            if (k0 + 64 < K) {        // L2 prefetch hint two steps ahead
                __builtin_prefetch(ga + k0 + 64, 0, 1);
                __builtin_prefetch(gw + k0 + 64, 0, 1);
            }
        }

        Frag16 af[4];
        #pragma unroll
        for (int mi = 0; mi < 4; ++mi) {
            const bf16* p = &As[(wm * 64 + mi * 16 + lr) * GSTRIDE + akb];
            af[mi].q[0] = *(const uint4*)p;
            af[mi].q[1] = *(const uint4*)(p + 16);
        }
        Frag16 bfr[2];
        #pragma unroll
        for (int ni = 0; ni < 2; ++ni) {
            const bf16* p = &Ws[(wn * 32 + ni * 16 + lr) * GSTRIDE + bkb];
            bfr[ni].q[0] = *(const uint4*)p;
            bfr[ni].q[1] = *(const uint4*)(p + 16);
        }
        #pragma unroll
        for (int mi = 0; mi < 4; ++mi)
            #pragma unroll
            for (int ni = 0; ni < 2; ++ni)
                acc[mi][ni] = wmma_bf16(af[mi].v, bfr[ni].v, acc[mi][ni]);
    }

    #pragma unroll
    for (int mi = 0; mi < 4; ++mi)
        #pragma unroll
        for (int ni = 0; ni < 2; ++ni)
            #pragma unroll
            for (int r = 0; r < 8; ++r) {
                int row = bm + wm * 64 + mi * 16 + r + 8 * half;
                int col = bn + wn * 32 + ni * 16 + lr;
                if (OUTBF16)
                    ((bf16*)Cout)[(size_t)row * LDC + col] = (bf16)acc[mi][ni][r];
                else
                    ((float*)Cout)[(size_t)row * LDC + col] = acc[mi][ni][r];
            }
}

// ---------------------------------------------------------------------------
// RoPE in-place on bf16 (B, L, nheads, 128); pairs (2i,2i+1); cos/sin[L,64].
// postscale folds the attention 1/sqrt(HD) into Q (1.0 for K).
// ---------------------------------------------------------------------------
__global__ void rope_bf16(bf16* __restrict__ t,
                          const float* __restrict__ cosb,
                          const float* __restrict__ sinb,
                          int nheads, int total_pairs, float postscale)
{
    int idx = blockIdx.x * blockDim.x + threadIdx.x;
    if (idx >= total_pairs) return;
    int i    = idx & 63;
    int rest = idx >> 6;
    int h    = rest % nheads;
    int bl   = rest / nheads;          // b*2048 + l
    int l    = bl & 2047;
    float c = cosb[l * 64 + i];
    float s = sinb[l * 64 + i];
    unsigned* p = (unsigned*)&t[((size_t)bl * nheads + h) * 128 + 2 * i];
    unsigned pk = *p;
    float tr = (float)__builtin_bit_cast(bf16, (unsigned short)(pk & 0xffff));
    float ti = (float)__builtin_bit_cast(bf16, (unsigned short)(pk >> 16));
    *p = pack_bf16x2((tr * c - ti * s) * postscale,
                     (tr * s + ti * c) * postscale);
}

// ---------------------------------------------------------------------------
// V transpose: (B, L, HKV, 128) bf16 -> (B*HKV, 128, L) bf16, LDS-tiled 64x64.
// grid = (L/64, B*HKV * 2).
// ---------------------------------------------------------------------------
__global__ __launch_bounds__(256) void transpose_v(const bf16* __restrict__ vb,
                                                   bf16* __restrict__ vt)
{
    __shared__ bf16 tile[64 * 72];
    const int tid  = threadIdx.x;
    const int l0   = blockIdx.x * 64;
    const int bkvh = blockIdx.y >> 1;          // b*8 + kvh
    const int d0   = (blockIdx.y & 1) * 64;
    const int b    = bkvh >> 3;
    const int kvh  = bkvh & 7;

    #pragma unroll
    for (int i = 0; i < 2; ++i) {
        int slot = tid + i * 256;              // 64 rows x 8 chunks
        int row = slot >> 3, ch = slot & 7;
        const bf16* g = vb + ((size_t)(b * 2048 + l0 + row) * 8 + kvh) * 128 + d0 + ch * 8;
        *(uint4*)&tile[row * 72 + ch * 8] = *(const uint4*)g;
    }
    __syncthreads();
    #pragma unroll
    for (int i = 0; i < 2; ++i) {
        int slot = tid + i * 256;              // 64 d-rows x 8 chunks
        int d = slot >> 3, ch = slot & 7;
        unsigned o[4];
        #pragma unroll
        for (int j = 0; j < 4; ++j) {
            bf16 a  = tile[(ch * 8 + 2 * j    ) * 72 + d];
            bf16 b2 = tile[(ch * 8 + 2 * j + 1) * 72 + d];
            o[j] = (unsigned)__builtin_bit_cast(unsigned short, a) |
                   ((unsigned)__builtin_bit_cast(unsigned short, b2) << 16);
        }
        uint4 o4; o4.x = o[0]; o4.y = o[1]; o4.z = o[2]; o4.w = o[3];
        *(uint4*)&vt[((size_t)bkvh * 128 + d0 + d) * 2048 + l0 + ch * 8] = o4;
    }
}

// ---------------------------------------------------------------------------
// Flash attention, causal, GQA repeat=4, all-bf16 operands (Q pre-scaled).
// grid = (L/128, B*H); 256 threads = 8 waves, each wave owns 16 Q rows.
// K and V^T tiles staged with async global->LDS 16B copies.
// ---------------------------------------------------------------------------
#define KST 136    // K-tile LDS row stride (128 + 8)
#define VST 72     // Vt LDS row stride (64 + 8)
#define PST 72     // P LDS row stride

__global__ __launch_bounds__(256) void flash_attn(
    const bf16* __restrict__ q,    // (B, L, 32*128) roped + pre-scaled bf16
    const bf16* __restrict__ k,    // (B, L, 8*128) roped bf16
    const bf16* __restrict__ vt,   // (B*8, 128, L) bf16 (pre-transposed)
    float* __restrict__ o)         // (B, L, 32*128) fp32
{
    __shared__ bf16 Ks[64 * KST];       // 64 keys x 128 d
    __shared__ bf16 Vt[128 * VST];      // d x 64 keys
    __shared__ bf16 Ps[8 * 16 * PST];   // per-wave 16 x 64 probs

    const int L = 2048;
    const int tid  = threadIdx.x;
    const int lane = tid & 31;
    const int w    = tid >> 5;
    const int lr   = lane & 15;
    const int half = lane >> 4;
    const int mtile = blockIdx.x;
    const int bh    = blockIdx.y;
    const int b     = bh >> 5;
    const int h     = bh & 31;
    const int kvh   = h >> 2;
    const int bkvh  = b * 8 + kvh;
    const int m0    = mtile * 128 + w * 16;

    // Q fragments resident in registers (16 x 128 per wave), direct bf16 loads
    Frag16 qf[4];
    {
        const bf16* qrow = q + ((size_t)(b * L + m0 + lr)) * 4096 + h * 128;
        const int akb = half ? 8 : 0;
        #pragma unroll
        for (int ks = 0; ks < 4; ++ks) {
            qf[ks].q[0] = *(const uint4*)&qrow[ks * 32 + akb];
            qf[ks].q[1] = *(const uint4*)&qrow[ks * 32 + akb + 16];
        }
    }

    const v8f zero8 = {0.f,0.f,0.f,0.f,0.f,0.f,0.f,0.f};
    v8f oacc[8];
    #pragma unroll
    for (int d = 0; d < 8; ++d) oacc[d] = zero8;
    float mrun[8], lrun[8];
    #pragma unroll
    for (int r = 0; r < 8; ++r) { mrun[r] = -3.0e38f; lrun[r] = 0.f; }

    const int ktend = mtile * 2 + 2;            // causal tile bound
    const int bkb = half ? 8 : 0;

    for (int kt = 0; kt < ktend; ++kt) {
        __syncthreads();
        // K tile: 64 rows x 128 bf16 = 1024 x 16B async copies
        #pragma unroll
        for (int i = 0; i < 4; ++i) {
            int slot = tid + i * 256;
            int row = slot >> 4, c16 = slot & 15;
            const bf16* g = k + ((size_t)(b * L + kt * 64 + row)) * 1024
                              + kvh * 128 + c16 * 8;
            async_b128(g, &Ks[row * KST + c16 * 8]);
        }
        // V^T tile: 128 d-rows x 64 bf16 = 1024 x 16B async copies
        #pragma unroll
        for (int i = 0; i < 4; ++i) {
            int slot = tid + i * 256;
            int d = slot >> 3, ch = slot & 7;
            const bf16* g = vt + ((size_t)bkvh * 128 + d) * 2048 + kt * 64 + ch * 8;
            async_b128(g, &Vt[d * VST + ch * 8]);
        }
        wait_async0();
        __syncthreads();

        // S = Q K^T (16 x 64 per wave); Q carries the 1/sqrt(HD) scale
        v8f s[4];
        #pragma unroll
        for (int ni = 0; ni < 4; ++ni) s[ni] = zero8;
        #pragma unroll
        for (int ks = 0; ks < 4; ++ks) {
            #pragma unroll
            for (int ni = 0; ni < 4; ++ni) {
                const bf16* p = &Ks[(ni * 16 + lr) * KST + ks * 32 + bkb];
                Frag16 kb;
                kb.q[0] = *(const uint4*)p;
                kb.q[1] = *(const uint4*)(p + 16);
                s[ni] = wmma_bf16(qf[ks].v, kb.v, s[ni]);
            }
        }

        // causal mask: only the diagonal tiles need it (wave-uniform branch,
        // no barriers inside -> legal divergence between waves)
        if (kt * 64 + 63 > m0) {
            #pragma unroll
            for (int ni = 0; ni < 4; ++ni) {
                int gcol = kt * 64 + ni * 16 + lr;
                #pragma unroll
                for (int r = 0; r < 8; ++r) {
                    int grow = m0 + r + 8 * half;
                    if (gcol > grow) s[ni][r] = -3.0e38f;
                }
            }
        }

        // online softmax
        float alpha[8];
        #pragma unroll
        for (int r = 0; r < 8; ++r) {
            float mx = mrun[r];
            #pragma unroll
            for (int ni = 0; ni < 4; ++ni)
                mx = fmaxf(mx, s[ni][r]);
            #pragma unroll
            for (int off = 1; off < 16; off <<= 1)
                mx = fmaxf(mx, __shfl_xor(mx, off, 32));
            alpha[r] = __expf(mrun[r] - mx);
            mrun[r]  = mx;
            float rs = 0.f;
            #pragma unroll
            for (int ni = 0; ni < 4; ++ni) {
                float pv = __expf(s[ni][r] - mx);
                rs += pv;
                Ps[(w * 16 + r + 8 * half) * PST + ni * 16 + lr] = (bf16)pv;
            }
            #pragma unroll
            for (int off = 1; off < 16; off <<= 1)
                rs += __shfl_xor(rs, off, 32);
            lrun[r] = lrun[r] * alpha[r] + rs;
        }
        #pragma unroll
        for (int d = 0; d < 8; ++d)
            #pragma unroll
            for (int r = 0; r < 8; ++r)
                oacc[d][r] *= alpha[r];

        __syncthreads();

        // O += P V
        const int pkb = half ? 8 : 0;
        #pragma unroll
        for (int ks = 0; ks < 2; ++ks) {
            Frag16 pf;
            const bf16* pp = &Ps[(w * 16 + lr) * PST + ks * 32 + pkb];
            pf.q[0] = *(const uint4*)pp;
            pf.q[1] = *(const uint4*)(pp + 16);
            #pragma unroll
            for (int d = 0; d < 8; ++d) {
                const bf16* vp = &Vt[(d * 16 + lr) * VST + ks * 32 + bkb];
                Frag16 vb;
                vb.q[0] = *(const uint4*)vp;
                vb.q[1] = *(const uint4*)(vp + 16);
                oacc[d] = wmma_bf16(pf.v, vb.v, oacc[d]);
            }
        }
    }

    // normalize and write (B, L, h*128 + d) fp32
    #pragma unroll
    for (int d = 0; d < 8; ++d) {
        #pragma unroll
        for (int r = 0; r < 8; ++r) {
            int grow = m0 + r + 8 * half;
            float val = oacc[d][r] / lrun[r];
            o[((size_t)(b * L + grow)) * 4096 + h * 128 + d * 16 + lr] = val;
        }
    }
}

// ---------------------------------------------------------------------------
extern "C" void kernel_launch(void* const* d_in, const int* in_sizes, int n_in,
                              void* d_out, int out_size, void* d_ws, size_t ws_size,
                              hipStream_t stream)
{
    (void)in_sizes; (void)n_in; (void)out_size;
    const float* x    = (const float*)d_in[0];
    const float* wq   = (const float*)d_in[1];
    const float* wk   = (const float*)d_in[2];
    const float* wv   = (const float*)d_in[3];
    const float* wo   = (const float*)d_in[4];
    const float* cosb = (const float*)d_in[5];
    const float* sinb = (const float*)d_in[6];
    float* out = (float*)d_out;

    // workspace: qb(32MB) kb(8MB) vb(8MB) vt(8MB) bf16 + ao(64MB) fp32 = 120MB
    char* wsp = (char*)d_ws;
    bf16* qb = (bf16*)wsp;                 wsp += (size_t)4096 * 4096 * 2;
    bf16* kb = (bf16*)wsp;                 wsp += (size_t)4096 * 1024 * 2;
    bf16* vb = (bf16*)wsp;                 wsp += (size_t)4096 * 1024 * 2;
    bf16* vt = (bf16*)wsp;                 wsp += (size_t)4096 * 1024 * 2;
    float* ao = (float*)wsp;
    if (ws_size < (size_t)(32 + 8 + 8 + 8 + 64) * 1024 * 1024) return;

    dim3 blk(256);

    // QKV projections (M = B*L = 4096, K = 4096), bf16 outputs
    gemm_xwT<4096, 4096, 4096, true><<<dim3(32, 32), blk, 0, stream>>>(x, wq, qb, 4096);
    gemm_xwT<4096, 4096, 1024, true><<<dim3(8,  32), blk, 0, stream>>>(x, wk, kb, 4096);
    gemm_xwT<4096, 4096, 1024, true><<<dim3(8,  32), blk, 0, stream>>>(x, wv, vb, 4096);

    // RoPE on q (32 heads, folds in 1/sqrt(HD)) and k (8 heads), in-place bf16
    {
        int qpairs = 2 * 2048 * 32 * 64;
        rope_bf16<<<qpairs / 256, 256, 0, stream>>>(qb, cosb, sinb, 32, qpairs,
                                                    0.08838834764831845f);
        int kpairs = 2 * 2048 * 8 * 64;
        rope_bf16<<<kpairs / 256, 256, 0, stream>>>(kb, cosb, sinb, 8, kpairs, 1.0f);
    }

    // V transpose to (B*HKV, 128, L)
    transpose_v<<<dim3(32, 32), blk, 0, stream>>>(vb, vt);

    // causal flash attention, GQA repeat=4
    flash_attn<<<dim3(16, 64), blk, 0, stream>>>(qb, kb, vt, ao);

    // output projection (fp32 out)
    gemm_xwT<4096, 4096, 4096, false><<<dim3(32, 32), blk, 0, stream>>>(ao, wo, out, 4096);
}
